// CFConv_40218073760108
// MI455X (gfx1250) — matrix operand order repaired and checked by reading, not compiled, not run on hardware
//
#include <hip/hip_runtime.h>

// CDNA5 (gfx1250) fused CFConv, f16 WMMA (f32 accumulate) version.
// ws layout: [ hv : V*64 f32 ][ h : V*64 f32 ]

typedef _Float16 h16;
typedef __attribute__((ext_vector_type(4)))  _Float16 v4h;
typedef __attribute__((ext_vector_type(8)))  _Float16 v8h;
typedef __attribute__((ext_vector_type(16))) _Float16 v16h;
typedef __attribute__((ext_vector_type(8)))  float    v8f;

#define H 64
#define AW 72                  // LDS row stride in halves: 144B, 16B-aligned chunks,
                               // 36-dword row rotation -> conflict-free b128 frag loads
#define TILE_M 16
#define WAVES_PER_BLOCK 8      // 256 threads = 8 wave32
#define ROWS_PER_BLOCK (TILE_M * WAVES_PER_BLOCK)   // 128

// Shifted softplus via raw v_exp_f32/v_log_f32 (log2 domain).
// Argument of log2 is in [1,2]: no subnormal/inf guards needed, unlike __logf.
__device__ __forceinline__ float sspf(float x) {
    const float LOG2E = 1.44269504088896340736f;
    const float LN2   = 0.69314718055994530942f;
    float t = __builtin_amdgcn_exp2f(-fabsf(x) * LOG2E);  // 2^(-|x|*log2e) = e^-|x|
    float l = __builtin_amdgcn_logf(1.0f + t);            // log2(1+t), arg in [1,2]
    return fmaf(LN2, l, fmaxf(x, 0.0f) - LN2);
}

// Two contiguous 16B chunks -> one v16h WMMA fragment.
__device__ __forceinline__ v16h frag2(const h16* base, int off0, int off1) {
    v8h lo = *(const v8h*)(base + off0);
    v8h hi = *(const v8h*)(base + off1);
    return __builtin_shufflevector(lo, hi, 0, 1, 2, 3, 4, 5, 6, 7,
                                           8, 9, 10, 11, 12, 13, 14, 15);
}

// One wave: C(16x64,f32) += A(16x64,f16) * W(64x64,f16) using V_WMMA_F32_16X16X32_F16.
// lds_a rows = A rows; lds_wt rows = W columns (transposed), both stride AW halves.
// A frag (16x32): lane half h reads K in {k0+8h..+7} and {k0+16+8h..+7}.
// B frag (32x16): lane half h reads K in {k0+8h..+7} and {k0+16+8h..+7} from W^T row n.
__device__ __forceinline__ void wave_gemm_f16(const h16* lds_a, int aRowBase,
                                              const h16* lds_wt,
                                              int lane16, int half, v8f acc[4]) {
#pragma unroll
    for (int k0 = 0; k0 < H; k0 += 32) {
        const h16* arow = lds_a + (aRowBase + lane16) * AW + k0;
        v16h a = frag2(arow, 8 * half, 16 + 8 * half);
#pragma unroll
        for (int nt = 0; nt < 4; ++nt) {
            const h16* brow = lds_wt + (nt * 16 + lane16) * AW + k0;
            v16h b = frag2(brow, 8 * half, 16 + 8 * half);
            acc[nt] = __builtin_amdgcn_wmma_f32_16x16x32_f16(
                false, a, false, b, (short)0, acc[nt], false, false);
        }
    }
}

// Cooperative stage of rows x 64 f32 from global -> f16 LDS tile (stride AW),
// clamping out-of-range rows. float4 global loads, v4h (8B) LDS stores.
__device__ __forceinline__ void stage_rows_f16(h16* lds, const float* __restrict__ g,
                                               int rows, int gRowBase, int gMaxRow) {
    int total = rows * 16;                        // float4 per row
    for (int idx = threadIdx.x; idx < total; idx += 256) {
        int r  = idx >> 4;
        int c4 = idx & 15;
        int gr = gRowBase + r;
        gr = gr < gMaxRow ? gr : gMaxRow - 1;
        float4 v = *((const float4*)(g + (size_t)gr * H) + c4);
        v4h p = { (h16)v.x, (h16)v.y, (h16)v.z, (h16)v.w };
        *(v4h*)(lds + r * AW + c4 * 4) = p;       // r*144 + c4*8 bytes, 8B aligned
    }
}

// Stage W[64][64] f32 row-major (k,n) -> transposed f16 LDS: lds[n*AW + k].
__device__ __forceinline__ void stage_wt_f16(h16* lds, const float* __restrict__ W) {
    for (int idx = threadIdx.x; idx < H * H; idx += 256) {
        int k = idx >> 6;
        int n = idx & 63;
        lds[n * AW + k] = (h16)W[idx];            // coalesced reads, tiny one-time cost
    }
}

__global__ void __launch_bounds__(256)
zero_kernel(float* __restrict__ p, int n) {
    int i = blockIdx.x * 256 + threadIdx.x;
    if (i < n) p[i] = 0.0f;
}

// Y[r,:] = (apply_ssp ? ssp : id)(X[r,:] @ W + bias)   for r in [0, R)
__global__ void __launch_bounds__(256)
proj_kernel(const float* __restrict__ X, const float* __restrict__ W,
            const float* __restrict__ bias, float* __restrict__ Y,
            int R, int apply_ssp) {
    __shared__ h16   lds_a[ROWS_PER_BLOCK * AW];   // 18432 B
    __shared__ h16   lds_wt[H * AW];               //  9216 B
    __shared__ float lds_b[H];

    int rowBlock = blockIdx.x * ROWS_PER_BLOCK;
    stage_rows_f16(lds_a, X, ROWS_PER_BLOCK, rowBlock, R);
    stage_wt_f16(lds_wt, W);
    if (threadIdx.x < H) lds_b[threadIdx.x] = bias[threadIdx.x];
    __syncthreads();

    int wave   = threadIdx.x >> 5;
    int lane   = threadIdx.x & 31;
    int lane16 = lane & 15;
    int half   = lane >> 4;
    int rowTile = rowBlock + wave * TILE_M;
    if (rowTile >= R) return;                  // whole-wave exit; no barriers below

    v8f acc[4] = {};
    wave_gemm_f16(lds_a, wave * TILE_M, lds_wt, lane16, half, acc);

#pragma unroll
    for (int nt = 0; nt < 4; ++nt) {
        int feat = nt * 16 + lane16;
        float bv = lds_b[feat];
#pragma unroll
        for (int i = 0; i < 8; ++i) {
            int row = rowTile + i + 8 * half;  // C layout: M = i + 8*half
            float v = acc[nt][i] + bv;
            if (apply_ssp) v = sspf(v);
            if (row < R) Y[(size_t)row * H + feat] = v;
        }
    }
}

// Fused edge pipeline: he = ssp(ssp(E@W1+b1)@W2+b2); m = hv[src]*he; h[dst] += m
__global__ void __launch_bounds__(256)
edge_kernel(const float* __restrict__ EF, const int* __restrict__ src,
            const int* __restrict__ dst,
            const float* __restrict__ W1, const float* __restrict__ b1,
            const float* __restrict__ W2, const float* __restrict__ b2,
            const float* __restrict__ hv, float* __restrict__ hacc, int E) {
    __shared__ h16   lds_a[ROWS_PER_BLOCK * AW];   // 18432 B
    __shared__ h16   lds_wt1[H * AW];              //  9216 B
    __shared__ h16   lds_wt2[H * AW];              //  9216 B
    __shared__ float lds_b1[H];
    __shared__ float lds_b2[H];

    int edgeBlock = blockIdx.x * ROWS_PER_BLOCK;
    stage_rows_f16(lds_a, EF, ROWS_PER_BLOCK, edgeBlock, E);
    stage_wt_f16(lds_wt1, W1);
    stage_wt_f16(lds_wt2, W2);
    if (threadIdx.x < H) {
        lds_b1[threadIdx.x] = b1[threadIdx.x];
        lds_b2[threadIdx.x] = b2[threadIdx.x];
    }
    __syncthreads();                // only barrier: everything below is wave-private

    int wave     = threadIdx.x >> 5;
    int lane     = threadIdx.x & 31;
    int lane16   = lane & 15;
    int half     = lane >> 4;
    int aRowBase = wave * TILE_M;

    // ---- layer 1: t1 = E_tile @ W1 ----
    v8f acc[4] = {};
    wave_gemm_f16(lds_a, aRowBase, lds_wt1, lane16, half, acc);

    // ssp(t1 + b1) -> f16, written back into this wave's own A rows
#pragma unroll
    for (int nt = 0; nt < 4; ++nt) {
        int feat = nt * 16 + lane16;
        float bv = lds_b1[feat];
#pragma unroll
        for (int i = 0; i < 8; ++i)
            lds_a[(aRowBase + i + 8 * half) * AW + feat] =
                (h16)sspf(acc[nt][i] + bv);
    }
    // no block barrier needed: rows are wave-private; in-wave LDS order is program order

    // ---- layer 2: t2 = ssp(t1) @ W2 ----
    v8f acc2[4] = {};
    wave_gemm_f16(lds_a, aRowBase, lds_wt2, lane16, half, acc2);

    // ---- epilogue: he = ssp(t2+b2); gather hv[src], scatter-add to h[dst] ----
    int edgeTile = edgeBlock + aRowBase;
#pragma unroll
    for (int nt = 0; nt < 4; ++nt) {
        int feat = nt * 16 + lane16;
        float bv = lds_b2[feat];
#pragma unroll
        for (int i = 0; i < 8; ++i) {
            int e  = edgeTile + i + 8 * half;
            int ec = e < E ? e : E - 1;
            int s  = src[ec];
            int d  = dst[ec];
            float he  = sspf(acc2[nt][i] + bv);
            float val = he * hv[(size_t)s * H + feat];   // L2-resident gather
            if (e < E) atomicAdd(&hacc[(size_t)d * H + feat], val);
        }
    }
}

extern "C" void kernel_launch(void* const* d_in, const int* in_sizes, int n_in,
                              void* d_out, int out_size, void* d_ws, size_t ws_size,
                              hipStream_t stream) {
    const float* node_feats = (const float*)d_in[0];
    const float* edge_feats = (const float*)d_in[1];
    const int*   src        = (const int*)d_in[2];
    const int*   dst        = (const int*)d_in[3];
    const float* Wn  = (const float*)d_in[4];
    const float* bn  = (const float*)d_in[5];
    const float* We1 = (const float*)d_in[6];
    const float* be1 = (const float*)d_in[7];
    const float* We2 = (const float*)d_in[8];
    const float* be2 = (const float*)d_in[9];
    const float* Wo  = (const float*)d_in[10];
    const float* bo  = (const float*)d_in[11];
    float* out = (float*)d_out;

    int V = in_sizes[0] / H;
    int E = in_sizes[2];

    float* hv   = (float*)d_ws;                 // V*64 f32
    float* hacc = hv + (size_t)V * H;           // V*64 f32

    int nh = V * H;
    zero_kernel<<<(nh + 255) / 256, 256, 0, stream>>>(hacc, nh);

    int vBlocks = (V + ROWS_PER_BLOCK - 1) / ROWS_PER_BLOCK;
    proj_kernel<<<vBlocks, 256, 0, stream>>>(node_feats, Wn, bn, hv, V, 0);

    int eBlocks = (E + ROWS_PER_BLOCK - 1) / ROWS_PER_BLOCK;
    edge_kernel<<<eBlocks, 256, 0, stream>>>(edge_feats, src, dst,
                                             We1, be1, We2, be2, hv, hacc, E);

    proj_kernel<<<vBlocks, 256, 0, stream>>>(hacc, Wo, bo, out, V, 1);
}
